// Region_Proposal_Network_47450798686777
// MI455X (gfx1250) — matrix operand order repaired and checked by reading, not compiled, not run on hardware
//
#include <hip/hip_runtime.h>

// ---------------- types for WMMA ----------------
typedef __attribute__((ext_vector_type(16))) __bf16 v16bf;
typedef __attribute__((ext_vector_type(8)))  float  v8f;
union ABfrag { v16bf v; uint4 q[2]; };

// ---------------- TDM availability ----------------
#if defined(__has_builtin)
#if __has_builtin(__builtin_amdgcn_tensor_load_to_lds) && __has_builtin(__builtin_amdgcn_s_wait_tensorcnt)
#define USE_TDM 1
#endif
#endif
#ifndef USE_TDM
#define USE_TDM 0
#endif

#if USE_TDM
typedef __attribute__((ext_vector_type(4))) unsigned int v4u;
typedef __attribute__((ext_vector_type(8))) int v8i;
typedef __attribute__((ext_vector_type(4))) int v4i;
#endif

// ---------------- problem constants ----------------
#define NIMG   4
#define CIN    512
#define COUT   512
#define HH     64
#define WW     64
#define NPIX   4096
#define NA     9
#define ATOT   36864      // NPIX*NA
#define KTOT   4608       // CIN*9
#define PRE_NMS  3000
#define POST_NMS 300
#define SORTN  65536
#define NWORDS 94         // ceil(3000/32)
#define NMS_THRESH 0.7f
#define MIN_SIZE   16.0f

// ---------------- workspace byte offsets ----------------
#define WT_OFF    0ull           // 9*512*512 bf16           = 4,718,592
#define XPAD_OFF  4718592ull     // 4*512*66*66 bf16         = 17,842,176
#define HID_OFF   22560768ull    // 4*512*4096 f32           = 33,554,432
#define BOX_OFF   56115200ull    // 4*36864*4 f32            = 2,359,296
#define SCORE_OFF 58474496ull    // 4*36864 f32              = 589,824
#define SKEY_OFF  59064320ull    // 4*65536 u64              = 2,097,152
#define BSORT_OFF 61161472ull    // 4*3000 float4            = 192,000
#define SUP0_OFF  61353472ull    // 4*3000 i32               = 48,000
#define MASK_OFF  61401472ull    // 4*3000*94 u32            = 4,512,000
#define KEEP_OFF  65913472ull    // 4*300 i32                = 4,800
#define KCNT_OFF  65918272ull    // 4 i32

// ---------------- d_out element offsets (float) ----------------
#define LOC_OUT   0ull           // 4*36864*4 = 589824
#define SCORE_OUT 589824ull      // 4*36864*2 = 294912
#define ROIS_OUT  884736ull      // 1200*4    = 4800
#define RIDX_OUT  889536ull      // 1200
#define ANCH_OUT  890736ull      // 36864*4   = 147456

// ---------------- helpers ----------------
__device__ __forceinline__ unsigned short f2bf(float f) {
  unsigned int u = __builtin_bit_cast(unsigned int, f);
  u += 0x7FFFu + ((u >> 16) & 1u);   // round-to-nearest-even
  return (unsigned short)(u >> 16);
}
__device__ __forceinline__ unsigned int fkey(float f) {
  unsigned int b = __builtin_bit_cast(unsigned int, f);
  return (b & 0x80000000u) ? ~b : (b | 0x80000000u);  // monotone (ascending) map
}

// ============ 1) W1 [512,512,3,3] f32 -> Wt [9][512][512] bf16 ============
__global__ __launch_bounds__(256) void convert_w_kernel(const float* __restrict__ W1,
                                                        unsigned short* __restrict__ Wt) {
  int o = blockIdx.x * 256 + threadIdx.x;          // < 9*512*512
  int ci = o & 511;
  int co = (o >> 9) & 511;
  int r  = o >> 18;
  Wt[o] = f2bf(W1[co * KTOT + ci * 9 + r]);
}

// ============ 2) x [4,512,64,64] f32 -> xpad [4][512][66][66] bf16 ============
__global__ __launch_bounds__(256) void pad_x_kernel(const float* __restrict__ x,
                                                    unsigned short* __restrict__ xpad) {
  int o = blockIdx.x * 256 + threadIdx.x;          // < 4*512*66*66
  int t1  = o / 66;
  int col = o - t1 * 66;
  int row = t1 % 66;
  int nc  = t1 / 66;                               // n*512 + c
  float v = 0.0f;
  if (row >= 1 && row <= 64 && col >= 1 && col <= 64)
    v = x[(nc * 64 + (row - 1)) * 64 + (col - 1)];
  xpad[o] = f2bf(v);
}

// ============ 3) 3x3 conv as implicit GEMM with bf16 WMMA ============
#define APITCH 40
#define BPITCH 40
#define WMMA_BF16(A, B, C) \
  __builtin_amdgcn_wmma_f32_16x16x32_bf16(false, (A).v, false, (B).v, (short)0, (C), false, false)

__global__ __launch_bounds__(256) void conv3_kernel(const unsigned short* __restrict__ Wt,
                                                    const unsigned short* __restrict__ xpad,
                                                    const float* __restrict__ b1,
                                                    float* __restrict__ hidden) {
  const int co0  = blockIdx.x * 128;               // 4 blocks of 128 output channels
  const int n    = blockIdx.y >> 6;                // image
  const int y    = blockIdx.y & 63;                // output row
  const int tid  = threadIdx.x;
  const int lane = tid & 31;
  const int wv   = tid >> 5;                       // 8 waves
  const int waveM = wv >> 1;                       // 0..3 (32-row subtiles)
  const int waveN = wv & 1;                        // 0..1 (32-col subtiles)

  __shared__ alignas(16) unsigned short Atile[128 * APITCH];  // [co][k], pitch 40
  __shared__ alignas(16) unsigned short Btile[64 * BPITCH];   // [x][k] (N-major)

  v8f c00 = {}, c01 = {}, c10 = {}, c11 = {};

  const int lrow = lane & 15;
  const int hi   = lane >> 4;

  for (int r = 0; r < 9; ++r) {
    const int kh = r / 3, kw = r - kh * 3;
    const unsigned short* wsrc = Wt + (size_t)r * (512 * 512) + (size_t)co0 * 512;
    const unsigned short* xrow0 = xpad + (((size_t)(n * 512) * 66) + (y + kh)) * 66 + kw;

    for (int ci0 = 0; ci0 < 512; ci0 += 32) {
      // ---- stage A tile: 128 rows x 32 k ----
#if USE_TDM
      if (wv == 0) {
        // Tensor Data Mover: 2D tile 32 (dim0) x 128 (dim1) of 2-byte elements,
        // row stride 512, LDS padding 4 DWORDs after every 16 DWORDs -> pitch 40.
        unsigned long long ga = (unsigned long long)(size_t)wsrc + (size_t)ci0 * 2;
        unsigned lds_addr = (unsigned)(size_t)(void*)Atile;
        v4u g0;
        g0.x = 1u;                                             // count=1 (valid), user mode
        g0.y = lds_addr;                                       // lds_addr [63:32]
        g0.z = (unsigned)(ga & 0xFFFFFFFFu);                   // global_addr lo
        g0.w = (unsigned)((ga >> 32) & 0x01FFFFFFu) | 0x80000000u;  // addr hi | type=2
        v8i g1;
        g1[0] = (int)0x06D10000u;  // data_size=2B | pad_enable | pad_interval=16dw | pad_amount=4dw
        g1[1] = (int)(512u << 16); // tensor_dim0 = 512 (lo16 in bits[63:48])
        g1[2] = (int)(512u << 16); // tensor_dim0 hi=0 | tensor_dim1 = 512 (lo16)
        g1[3] = (int)(32u << 16);  // tensor_dim1 hi=0 | tile_dim0 = 32
        g1[4] = 128;               // tile_dim1 = 128 | tile_dim2 = 0
        g1[5] = 512;               // tensor_dim0_stride lo32 = 512
        g1[6] = (int)(512u << 16); // stride0 hi=0 | tensor_dim1_stride lo16 = 512
        g1[7] = 0;                 // tensor_dim1_stride hi = 0
        v4i z4 = {0, 0, 0, 0};
        v8i z8 = {0, 0, 0, 0, 0, 0, 0, 0};
        // 6-arg form (this toolchain): (g0, g1, g2, g3, g4, cpol)
        __builtin_amdgcn_tensor_load_to_lds(g0, g1, z4, z4, z8, 0);
        __builtin_amdgcn_s_wait_tensorcnt(0);
      }
#else
      {
        int c0 = tid * 2;
#pragma unroll
        for (int q = 0; q < 2; ++q) {
          int cc  = c0 + q;                        // 0..511
          int row = cc >> 2, seg = cc & 3;
          uint4 v = *(const uint4*)(wsrc + (size_t)row * 512 + ci0 + seg * 8);
          *(uint4*)(Atile + row * APITCH + seg * 8) = v;
        }
      }
#endif
      // ---- stage B tile: 32 k rows x 64 pixels, stored N-major ----
      {
        int i = tid >> 3;                          // k row 0..31
        int j = tid & 7;                           // x chunk
        const unsigned short* src = xrow0 + (size_t)(ci0 + i) * (66 * 66) + j * 8;
#pragma unroll
        for (int e = 0; e < 8; ++e)
          Btile[(j * 8 + e) * BPITCH + i] = src[e];
        if (ci0 + 32 < 512)                        // prefetch next ci-step's B rows
          __builtin_prefetch(src + (size_t)32 * (66 * 66), 0, 1);
      }
      __syncthreads();

      // ---- load fragments per documented 16-bit layouts ----
      ABfrag a0, a1, b0, b1v;
      {
        const unsigned short* ar0 = Atile + (waveM * 32 + lrow) * APITCH + hi * 8;
        a0.q[0] = *(const uint4*)(ar0);            // K = hi*8 .. +7
        a0.q[1] = *(const uint4*)(ar0 + 16);       // K = hi*8+16 .. +23
        const unsigned short* ar1 = Atile + (waveM * 32 + 16 + lrow) * APITCH + hi * 8;
        a1.q[0] = *(const uint4*)(ar1);
        a1.q[1] = *(const uint4*)(ar1 + 16);
        const unsigned short* br0 = Btile + (waveN * 32 + lrow) * BPITCH + hi * 8;
        b0.q[0] = *(const uint4*)(br0);            // K = hi*8 .. +7
        b0.q[1] = *(const uint4*)(br0 + 16);       // K = hi*8+16 .. +23
        const unsigned short* br1 = Btile + (waveN * 32 + 16 + lrow) * BPITCH + hi * 8;
        b1v.q[0] = *(const uint4*)(br1);
        b1v.q[1] = *(const uint4*)(br1 + 16);
      }
      c00 = WMMA_BF16(a0, b0, c00);
      c01 = WMMA_BF16(a0, b1v, c01);
      c10 = WMMA_BF16(a1, b0, c10);
      c11 = WMMA_BF16(a1, b1v, c11);
      __syncthreads();
    }
  }

  // ---- epilogue: bias + ReLU, store f32 hidden[n][co][y][x] ----
  const int colbase = waveN * 32 + lrow;
  const int rowoff  = hi * 8;
#pragma unroll
  for (int mi = 0; mi < 2; ++mi) {
#pragma unroll
    for (int ni = 0; ni < 2; ++ni) {
      v8f cc = (mi == 0) ? (ni == 0 ? c00 : c01) : (ni == 0 ? c10 : c11);
      int xcol = colbase + ni * 16;
#pragma unroll
      for (int rr = 0; rr < 8; ++rr) {
        int co = co0 + waveM * 32 + mi * 16 + rowoff + rr;
        float v = cc[rr] + b1[co];
        hidden[((size_t)(n * 512 + co) << 12) + (y << 6) + xcol] = fmaxf(v, 0.0f);
      }
    }
  }
}

// ============ 4) head: 1x1 convs + outputs + decode/clip/filter ============
__global__ __launch_bounds__(256) void head_kernel(const float* __restrict__ hidden,
                                                   const float* __restrict__ Ws, const float* __restrict__ bs,
                                                   const float* __restrict__ Wl, const float* __restrict__ bl,
                                                   const int* __restrict__ img_h_p, const int* __restrict__ img_w_p,
                                                   float* __restrict__ out,
                                                   float* __restrict__ boxes_all, float* __restrict__ score_all) {
  const int n  = blockIdx.x >> 8;
  const int g  = blockIdx.x & 255;
  const int p0 = g * 16;
  const int tid = threadIdx.x;
  __shared__ float hbuf[512 * 16];                 // [c][px]
  __shared__ float obuf[16 * 56];                  // [px][o] (54 used)

  for (int idx = tid; idx < 512 * 16; idx += 256) {
    int c = idx >> 4, px = idx & 15;
    hbuf[idx] = hidden[((size_t)(n * 512 + c) << 12) + p0 + px];
  }
  __syncthreads();

  for (int idx = tid; idx < 54 * 16; idx += 256) {
    int o = idx % 54, px = idx / 54;
    const float* w;
    float acc;
    if (o < 36) { w = Wl + o * 512; acc = bl[o]; }
    else        { w = Ws + (o - 36) * 512; acc = bs[o - 36]; }
    for (int c = 0; c < 512; ++c) acc += w[c] * hbuf[c * 16 + px];
    obuf[px * 56 + o] = acc;
  }
  __syncthreads();

  const float fih = (float)(*img_h_p);
  const float fiw = (float)(*img_w_p);
  for (int idx = tid; idx < 16 * NA; idx += 256) {
    int px = idx / NA, a = idx - px * NA;
    int p  = p0 + px;
    int yy = p >> 6, xx = p & 63;
    int ri = a / 3, si = a - ri * 3;
    float ratio = (ri == 0) ? 0.5f : ((ri == 1) ? 1.0f : 2.0f);
    float scale = (si == 0) ? 8.0f : ((si == 1) ? 16.0f : 32.0f);
    float sr = sqrtf(ratio);
    float bh = 16.0f * scale * sr;
    float bw = 16.0f * scale / sr;
    float ay = yy * 16.0f + 8.0f, ax = xx * 16.0f + 8.0f;
    float a0 = ay - 0.5f * bh, a1 = ax - 0.5f * bw, a2 = ay + 0.5f * bh, a3 = ax + 0.5f * bw;
    int aidx = p * NA + a;
    if (n == 0) {
      float* ao = out + ANCH_OUT + (size_t)aidx * 4;
      ao[0] = a0; ao[1] = a1; ao[2] = a2; ao[3] = a3;
    }
    float dy = obuf[px * 56 + 4 * a + 0];
    float dx = obuf[px * 56 + 4 * a + 1];
    float dh = obuf[px * 56 + 4 * a + 2];
    float dw = obuf[px * 56 + 4 * a + 3];
    float s0 = obuf[px * 56 + 36 + 2 * a];
    float s1 = obuf[px * 56 + 36 + 2 * a + 1];

    float* lo = out + LOC_OUT + ((size_t)n * ATOT + aidx) * 4;
    lo[0] = dy; lo[1] = dx; lo[2] = dh; lo[3] = dw;
    float* so = out + SCORE_OUT + ((size_t)n * ATOT + aidx) * 2;
    so[0] = s0; so[1] = s1;

    float ah = a2 - a0, aw = a3 - a1;
    float cy = a0 + 0.5f * ah, cx = a1 + 0.5f * aw;
    float ncy = dy * ah + cy, ncx = dx * aw + cx;
    float nh = expf(dh) * ah, nw = expf(dw) * aw;
    float r0 = fminf(fmaxf(ncy - 0.5f * nh, 0.0f), fih);
    float r1 = fminf(fmaxf(ncx - 0.5f * nw, 0.0f), fiw);
    float r2 = fminf(fmaxf(ncy + 0.5f * nh, 0.0f), fih);
    float r3 = fminf(fmaxf(ncx + 0.5f * nw, 0.0f), fiw);
    bool ok = ((r2 - r0) >= MIN_SIZE) && ((r3 - r1) >= MIN_SIZE);
    float fg = 1.0f / (1.0f + expf(s0 - s1));      // softmax[...,1]
    score_all[(size_t)n * ATOT + aidx] = ok ? fg : -__builtin_inff();
    float* bp = boxes_all + ((size_t)n * ATOT + aidx) * 4;
    bp[0] = r0; bp[1] = r1; bp[2] = r2; bp[3] = r3;
  }
}

// ============ 5) selection: pack keys, per-image bitonic sort, gather ============
__global__ __launch_bounds__(256) void sortprep_kernel(const float* __restrict__ score_all,
                                                       unsigned long long* __restrict__ skeys) {
  int g = blockIdx.x * 256 + threadIdx.x;          // < 4*65536
  int n = g >> 16, i = g & 65535;
  float s = (i < ATOT) ? score_all[(size_t)n * ATOT + i] : -__builtin_inff();
  skeys[g] = ((unsigned long long)(~fkey(s)) << 32) | (unsigned int)i;  // asc = score desc, idx asc
}

__global__ __launch_bounds__(1024) void bitonic_kernel(unsigned long long* __restrict__ skeys) {
  unsigned long long* a = skeys + ((size_t)blockIdx.x << 16);
  const int tid = threadIdx.x;
  for (int k = 2; k <= SORTN; k <<= 1) {
    for (int j = k >> 1; j > 0; j >>= 1) {
      for (int i = tid; i < SORTN; i += 1024) {
        int ixj = i ^ j;
        if (ixj > i) {
          unsigned long long va = a[i], vb = a[ixj];
          bool up = ((i & k) == 0);
          if ((va > vb) == up) { a[i] = vb; a[ixj] = va; }
        }
      }
      __syncthreads();
    }
  }
}

__global__ __launch_bounds__(256) void gather_kernel(const unsigned long long* __restrict__ skeys,
                                                     const float* __restrict__ boxes_all,
                                                     const float* __restrict__ score_all,
                                                     float4* __restrict__ boxes_sorted,
                                                     int* __restrict__ sup0) {
  int t = blockIdx.x * 256 + threadIdx.x;
  if (t >= NIMG * PRE_NMS) return;
  int n = t / PRE_NMS, i = t - n * PRE_NMS;
  unsigned long long K = skeys[((size_t)n << 16) + i];
  unsigned int idx = (unsigned int)K;
  float4 bx = make_float4(0.0f, 0.0f, 0.0f, 0.0f);
  int sup = 1;
  if (idx < ATOT) {
    const float* bp = boxes_all + ((size_t)n * ATOT + idx) * 4;
    bx = make_float4(bp[0], bp[1], bp[2], bp[3]);
    sup = (score_all[(size_t)n * ATOT + idx] == -__builtin_inff()) ? 1 : 0;
  }
  boxes_sorted[t] = bx;
  sup0[t] = sup;
}

// ============ 6) NMS: bitmask + sequential scan + emit ============
__global__ __launch_bounds__(96) void iou_mask_kernel(const float4* __restrict__ boxes_sorted,
                                                      unsigned int* __restrict__ mask) {
  const int i = blockIdx.x, n = blockIdx.y, j = threadIdx.x;
  if (j >= NWORDS) return;
  const float4* B = boxes_sorted + (size_t)n * PRE_NMS;
  float4 bi = B[i];
  float ai = fmaxf(bi.z - bi.x, 0.0f) * fmaxf(bi.w - bi.y, 0.0f);
  unsigned int bits = 0;
#pragma unroll 4
  for (int b = 0; b < 32; ++b) {
    int jj = j * 32 + b;
    if (jj < PRE_NMS && jj > i) {
      float4 bj = B[jj];
      float aj = fmaxf(bj.z - bj.x, 0.0f) * fmaxf(bj.w - bj.y, 0.0f);
      float yy1 = fmaxf(bi.x, bj.x), xx1 = fmaxf(bi.y, bj.y);
      float yy2 = fminf(bi.z, bj.z), xx2 = fminf(bi.w, bj.w);
      float inter = fmaxf(yy2 - yy1, 0.0f) * fmaxf(xx2 - xx1, 0.0f);
      float iou = inter / (ai + aj - inter + 1e-9f);
      if (iou > NMS_THRESH) bits |= (1u << b);
    }
  }
  mask[((size_t)n * PRE_NMS + i) * NWORDS + j] = bits;
}

__global__ __launch_bounds__(96) void nms_scan_kernel(const unsigned int* __restrict__ mask,
                                                      const int* __restrict__ sup0,
                                                      int* __restrict__ keeplist,
                                                      int* __restrict__ keepcnt) {
  const int n = blockIdx.x, tid = threadIdx.x;
  __shared__ unsigned int remv[NWORDS];
  __shared__ int cnt;
  if (tid < NWORDS) {
    unsigned int w = 0;
    for (int b = 0; b < 32; ++b) {
      int jj = tid * 32 + b;
      if (jj < PRE_NMS && sup0[(size_t)n * PRE_NMS + jj]) w |= (1u << b);
    }
    remv[tid] = w;
  }
  if (tid == 0) cnt = 0;
  __syncthreads();
  for (int i = 0; i < PRE_NMS; ++i) {
    bool alive = !((remv[i >> 5] >> (i & 31)) & 1u);
    __syncthreads();
    if (alive) {
      if (tid < NWORDS) remv[tid] |= mask[((size_t)n * PRE_NMS + i) * NWORDS + tid];
      if (tid == 0) {
        if (cnt < POST_NMS) keeplist[n * POST_NMS + cnt] = i;
        cnt++;
      }
    }
    __syncthreads();
  }
  if (tid == 0) keepcnt[n] = cnt;
}

__global__ __launch_bounds__(300) void write_rois_kernel(const float4* __restrict__ boxes_sorted,
                                                         const int* __restrict__ keeplist,
                                                         const int* __restrict__ keepcnt,
                                                         float* __restrict__ out) {
  const int n = blockIdx.x, s = threadIdx.x;   // s < 300
  int cnt = keepcnt[n];
  float4 bx = make_float4(0.0f, 0.0f, 0.0f, 0.0f);
  if (s < cnt) {
    int i = keeplist[n * POST_NMS + s];
    bx = boxes_sorted[(size_t)n * PRE_NMS + i];
  }
  float* rp = out + ROIS_OUT + ((size_t)n * POST_NMS + s) * 4;
  rp[0] = bx.x; rp[1] = bx.y; rp[2] = bx.z; rp[3] = bx.w;
  out[RIDX_OUT + (size_t)n * POST_NMS + s] = (float)n;
}

// ---------------- launcher ----------------
extern "C" void kernel_launch(void* const* d_in, const int* in_sizes, int n_in,
                              void* d_out, int out_size, void* d_ws, size_t ws_size,
                              hipStream_t stream) {
  const float* x    = (const float*)d_in[0];
  const int*   ihp  = (const int*)d_in[1];
  const int*   iwp  = (const int*)d_in[2];
  const float* W1   = (const float*)d_in[3];
  const float* b1   = (const float*)d_in[4];
  const float* Ws   = (const float*)d_in[5];
  const float* bs   = (const float*)d_in[6];
  const float* Wl   = (const float*)d_in[7];
  const float* bl   = (const float*)d_in[8];
  float* out = (float*)d_out;
  char*  ws  = (char*)d_ws;

  unsigned short* Wt     = (unsigned short*)(ws + WT_OFF);
  unsigned short* xpad   = (unsigned short*)(ws + XPAD_OFF);
  float*          hidden = (float*)(ws + HID_OFF);
  float*          boxesA = (float*)(ws + BOX_OFF);
  float*          scoreA = (float*)(ws + SCORE_OFF);
  unsigned long long* skeys = (unsigned long long*)(ws + SKEY_OFF);
  float4*         bsort  = (float4*)(ws + BSORT_OFF);
  int*            sup0   = (int*)(ws + SUP0_OFF);
  unsigned int*   mask   = (unsigned int*)(ws + MASK_OFF);
  int*            keepl  = (int*)(ws + KEEP_OFF);
  int*            keepc  = (int*)(ws + KCNT_OFF);

  convert_w_kernel<<<(9 * 512 * 512) / 256, 256, 0, stream>>>(W1, Wt);
  pad_x_kernel<<<(NIMG * CIN * 66 * 66) / 256, 256, 0, stream>>>(x, xpad);
  conv3_kernel<<<dim3(4, NIMG * HH), 256, 0, stream>>>(Wt, xpad, b1, hidden);
  head_kernel<<<NIMG * 256, 256, 0, stream>>>(hidden, Ws, bs, Wl, bl, ihp, iwp, out, boxesA, scoreA);
  sortprep_kernel<<<(NIMG * SORTN) / 256, 256, 0, stream>>>(scoreA, skeys);
  bitonic_kernel<<<NIMG, 1024, 0, stream>>>(skeys);
  gather_kernel<<<(NIMG * PRE_NMS + 255) / 256, 256, 0, stream>>>(skeys, boxesA, scoreA, bsort, sup0);
  iou_mask_kernel<<<dim3(PRE_NMS, NIMG), 96, 0, stream>>>(bsort, mask);
  nms_scan_kernel<<<NIMG, 96, 0, stream>>>(mask, sup0, keepl, keepc);
  write_rois_kernel<<<NIMG, 300, 0, stream>>>(bsort, keepl, keepc, out);
}
